// Encoder_17446157157163
// MI455X (gfx1250) — matrix operand order repaired and checked by reading, not compiled, not run on hardware
//
#include <hip/hip_runtime.h>
#include <hip/hip_bf16.h>

// ---------------------------------------------------------------------------
// 2-layer LayerNorm-LSTM on MI455X (gfx1250), wave32 + v_wmma_f32_16x16x32_f16.
//
// Latency-bound sequential scan (T=2048) -> 2 persistent workgroups x 16 batch
// rows (M=16 = WMMA tile height); all sync intra-workgroup. Packed f16 weights
// stay L2-resident. Cell state c in VGPRs; h states in LDS (A-fragments via
// ds_load_b128). Two-pass cell (a=x@W then r=h@U) with the LN'd a-term parked
// in a per-wave-private LDS z-buffer so only ONE 64-VGPR accumulator set is
// live at a time -> no scratch spills. kt loops kept rolled (#pragma unroll 1).
// global_prefetch_b8 of next k-tile's B-fragments hides L2 latency at loop
// boundaries (single-wave critical path, low occupancy by design).
// ---------------------------------------------------------------------------

typedef __attribute__((ext_vector_type(16))) _Float16 v16h;
typedef __attribute__((ext_vector_type(8)))  _Float16 v8h;
typedef __attribute__((ext_vector_type(8)))  float    v8f;

#define TT 2048
#define DD 64
#define HH 256
#define NHH 1024
#define LN_EPS 1e-3f

// packed-weight layout (f16): per 32(K)x16(N) tile, 512 elems; lane l holds
// n = l&15, K-range (l<16 ? 0..15 : 16..31), element e <-> k = e + 16*(l>>4).
// tile (kt,nt) of a matrix at (kt*64 + nt)*512.
#define W0P_OFF 0
#define U0P_OFF (65536)
#define W1P_OFF (65536 + 262144)
#define U1P_OFF (65536 + 2*262144)
#define PACK_TOTAL (65536 + 3*262144)   // 851968 f16 = 1.66 MB

__device__ __forceinline__ float sigf(float x) { return 1.f / (1.f + __expf(-x)); }

// A-fragment (16x32 f16, contiguous layout: lanes0-15 row M, e0..15=K k0..k0+15;
// lanes16-31 K offset +16) from LDS row-major buffer.
__device__ __forceinline__ v16h load_afrag(const _Float16* base, int stride, int k0, int lane) {
  int row  = lane & 15;
  int koff = k0 + ((lane & 16) ? 16 : 0);
  const _Float16* p = base + row * stride + koff;
  v8h lo = *(const v8h*)(p);
  v8h hi = *(const v8h*)(p + 8);
  v16h r;
#pragma unroll
  for (int i = 0; i < 8; ++i) { r[i] = lo[i]; r[i + 8] = hi[i]; }
  return r;
}

// 16-lane-half sum reduction + LDS row-stat combine.
// red: [8 waves][16 rows][2], stat: [16 rows][2] {mean, rstd}
__device__ __forceinline__ void row_stats(float* s, float* q,   // per-v partials (8 each)
                                          float* red, float* stat,
                                          float invN, int wave, int lane, int tid) {
#pragma unroll
  for (int off = 8; off >= 1; off >>= 1) {
#pragma unroll
    for (int v = 0; v < 8; ++v) { s[v] += __shfl_xor(s[v], off); q[v] += __shfl_xor(q[v], off); }
  }
  if ((lane & 15) == 0) {
    int rb = (lane >> 4) * 8;
#pragma unroll
    for (int v = 0; v < 8; ++v) {
      float* p = red + (wave * 16 + rb + v) * 2;
      p[0] = s[v]; p[1] = q[v];
    }
  }
  __syncthreads();
  if (tid < 16) {
    float s0 = 0, s1 = 0;
#pragma unroll
    for (int w = 0; w < 8; ++w) { const float* p = red + (w * 16 + tid) * 2; s0 += p[0]; s1 += p[1]; }
    float m = s0 * invN, vv = s1 * invN - m * m;
    stat[tid * 2 + 0] = m;
    stat[tid * 2 + 1] = rsqrtf(vv + LN_EPS);
  }
  __syncthreads();
}

// One LN-LSTM cell update for 16 rows. Wave `w` owns N-tiles {g*16+2w, g*16+2w+1}
// for gates g=0..3 (tile index ti: gate=ti>>1, sub=ti&1).
template<int KX>
__device__ void cell(const _Float16* __restrict__ shAx,   // LDS 16 x KX (feeds W)
                     const _Float16* __restrict__ shAh,   // LDS 16 x 256 (feeds U)
                     const _Float16* __restrict__ Wp,     // packed (KX/32 x 64 tiles)
                     const _Float16* __restrict__ Up,     // packed (8 x 64 tiles)
                     const float* __restrict__ gk, const float* __restrict__ bk,
                     const float* __restrict__ gr, const float* __restrict__ br,
                     const float* __restrict__ bias,
                     const float* __restrict__ gs, const float* __restrict__ bs,
                     const unsigned char* shMask,
                     float* red, float* stat,
                     _Float16* zbuf,       // LDS 64 tiles x 32 lanes x 8 (f16)
                     v8f c[2],
                     _Float16* outAct,     // unmasked h_new (layer0 -> layer1), may be null
                     _Float16* outState,   // masked h state
                     int wave, int lane, int tid) {
  const v8f vzero = {0.f,0.f,0.f,0.f,0.f,0.f,0.f,0.f};
  int hh   = (lane >> 4) * 8;   // row offset of this lane's accumulator half
  int ln15 = lane & 15;
  v8f acc[8];

  // ================= pass 1: a = x @ W, LN, park a' in LDS =================
#pragma unroll
  for (int i = 0; i < 8; ++i) acc[i] = vzero;
#pragma unroll 1
  for (int kt = 0; kt < KX / 32; ++kt) {
    // prefetch next k-tile's B-fragments (clamped; speculative, zero reg cost)
    int ktn = (kt + 1 < KX / 32) ? kt + 1 : kt;
    v16h a = load_afrag(shAx, KX, kt * 32, lane);
#pragma unroll
    for (int ti = 0; ti < 8; ++ti) {
      int ntile = (ti >> 1) * 16 + wave * 2 + (ti & 1);
      __builtin_prefetch(Wp + ((size_t)(ktn * 64 + ntile)) * 512 + lane * 16, 0, 3);
      const _Float16* bp = Wp + ((size_t)(kt * 64 + ntile)) * 512 + lane * 16;
      v16h b = *(const v16h*)bp;
      acc[ti] = __builtin_amdgcn_wmma_f32_16x16x32_f16(false, a, false, b,
                                                       (short)0, acc[ti], false, false);
    }
  }
  // pull pass-2's first k-tile toward the WGP while we do LN stats/barriers
#pragma unroll
  for (int ti = 0; ti < 8; ++ti) {
    int ntile = (ti >> 1) * 16 + wave * 2 + (ti & 1);
    __builtin_prefetch(Up + ((size_t)ntile) * 512 + lane * 16, 0, 3);
  }
  {
    float s[8], q[8];
#pragma unroll
    for (int v = 0; v < 8; ++v) {
      float a0 = 0, a1 = 0;
#pragma unroll
      for (int ti = 0; ti < 8; ++ti) { float xa = acc[ti][v]; a0 += xa; a1 += xa * xa; }
      s[v] = a0; q[v] = a1;
    }
    row_stats(s, q, red, stat, 1.f / NHH, wave, lane, tid);
  }
#pragma unroll
  for (int ti = 0; ti < 8; ++ti) {
    int g = ti >> 1, sub = ti & 1;
    int col = g * 256 + wave * 32 + sub * 16 + ln15;
    float gkc = gk[col], bkc = bk[col], bc = bias[col];
    v8h zv;
#pragma unroll
    for (int v = 0; v < 8; ++v) {
      int row = hh + v;
      float ap = (acc[ti][v] - stat[row * 2]) * stat[row * 2 + 1] * gkc + bkc + bc;
      zv[v] = (_Float16)ap;
    }
    *(v8h*)(zbuf + ((size_t)(wave * 8 + ti) * 32 + lane) * 8) = zv;
  }

  // ================= pass 2: r = h @ U, LN, gates =================
#pragma unroll
  for (int i = 0; i < 8; ++i) acc[i] = vzero;
#pragma unroll 1
  for (int kt = 0; kt < 8; ++kt) {
    int ktn = (kt + 1 < 8) ? kt + 1 : kt;
    v16h a = load_afrag(shAh, HH, kt * 32, lane);
#pragma unroll
    for (int ti = 0; ti < 8; ++ti) {
      int ntile = (ti >> 1) * 16 + wave * 2 + (ti & 1);
      __builtin_prefetch(Up + ((size_t)(ktn * 64 + ntile)) * 512 + lane * 16, 0, 3);
      const _Float16* bp = Up + ((size_t)(kt * 64 + ntile)) * 512 + lane * 16;
      v16h b = *(const v16h*)bp;
      acc[ti] = __builtin_amdgcn_wmma_f32_16x16x32_f16(false, a, false, b,
                                                       (short)0, acc[ti], false, false);
    }
  }
  {
    float s[8], q[8];
#pragma unroll
    for (int v = 0; v < 8; ++v) {
      float a0 = 0, a1 = 0;
#pragma unroll
      for (int ti = 0; ti < 8; ++ti) { float xr = acc[ti][v]; a0 += xr; a1 += xr * xr; }
      s[v] = a0; q[v] = a1;
    }
    row_stats(s, q, red, stat, 1.f / NHH, wave, lane, tid);
  }

  v8f cn[2], zo[2];
#pragma unroll
  for (int sub = 0; sub < 2; ++sub) {
#pragma unroll
    for (int v = 0; v < 8; ++v) {
      int row = hh + v;
      float mR = stat[row * 2], rR = stat[row * 2 + 1];
      float zg[4];
#pragma unroll
      for (int g = 0; g < 4; ++g) {
        int ti  = g * 2 + sub;
        int col = g * 256 + wave * 32 + sub * 16 + ln15;
        float ap = (float)zbuf[((size_t)(wave * 8 + ti) * 32 + lane) * 8 + v];
        zg[g] = ap + (acc[ti][v] - mR) * rR * gr[col] + br[col];
      }
      float iv = sigf(zg[0]), fv = sigf(zg[1]), gv = tanhf(zg[2]);
      cn[sub][v] = fv * c[sub][v] + iv * gv;
      zo[sub][v] = zg[3];
    }
  }

  // ---- LN over new cell state (256 per row) ----
  {
    float s[8], q[8];
#pragma unroll
    for (int v = 0; v < 8; ++v) {
      float x0 = cn[0][v], x1 = cn[1][v];
      s[v] = x0 + x1; q[v] = x0 * x0 + x1 * x1;
    }
    row_stats(s, q, red, stat, 1.f / HH, wave, lane, tid);
  }

#pragma unroll
  for (int sub = 0; sub < 2; ++sub) {
#pragma unroll
    for (int v = 0; v < 8; ++v) {
      int row = hh + v;
      int colH = wave * 32 + sub * 16 + ln15;
      float cN = (cn[sub][v] - stat[row * 2]) * stat[row * 2 + 1] * gs[colH] + bs[colH];
      float hN = sigf(zo[sub][v]) * tanhf(cN);
      bool mok = (shMask[row] != 0);
      c[sub][v] = mok ? cN : c[sub][v];
      if (outAct) outAct[row * HH + colH] = (_Float16)hN;
      if (mok) outState[row * HH + colH] = (_Float16)hN;
    }
  }
  __syncthreads();
}

// --- one-shot weight pack: f32 row-major -> f16 WMMA B-fragment tiles in ws ---
__global__ void pack_weights(const float* __restrict__ W0, const float* __restrict__ U0,
                             const float* __restrict__ W1, const float* __restrict__ U1,
                             _Float16* __restrict__ ws) {
  int idx = blockIdx.x * blockDim.x + threadIdx.x;
  if (idx >= PACK_TOTAL) return;
  int tile = idx >> 9, within = idx & 511;
  int lane = within >> 4, e = within & 15;
  int n = lane & 15;
  int k = e + ((lane >> 4) << 4);
  const float* src; int mt;
  if (tile < 128)       { src = W0; mt = tile; }
  else if (tile < 640)  { src = U0; mt = tile - 128; }
  else if (tile < 1152) { src = W1; mt = tile - 640; }
  else                  { src = U1; mt = tile - 1152; }
  int kt = mt >> 6, nt = mt & 63;
  ws[idx] = (_Float16)src[(size_t)(kt * 32 + k) * NHH + nt * 16 + n];
}

__global__ __launch_bounds__(256) void lstm_persist(
    const float* __restrict__ x, const unsigned char* __restrict__ mask,
    const _Float16* __restrict__ ws,
    const float* __restrict__ gk0, const float* __restrict__ bk0,
    const float* __restrict__ gr0, const float* __restrict__ br0,
    const float* __restrict__ b0v, const float* __restrict__ gs0, const float* __restrict__ bs0,
    const float* __restrict__ gk1, const float* __restrict__ bk1,
    const float* __restrict__ gr1, const float* __restrict__ br1,
    const float* __restrict__ b1v, const float* __restrict__ gs1, const float* __restrict__ bs1,
    const float* __restrict__ Wd, const float* __restrict__ bd,
    float* __restrict__ out) {
  __shared__ __align__(16) _Float16 sh_x[16 * DD];
  __shared__ __align__(16) _Float16 sh_h0[16 * HH];   // masked state, layer0
  __shared__ __align__(16) _Float16 sh_h0a[16 * HH];  // unmasked nh0 (feeds layer1)
  __shared__ __align__(16) _Float16 sh_h1[16 * HH];   // masked state, layer1
  __shared__ __align__(16) _Float16 zbuf[64 * 32 * 8]; // parked a' (32 KB)
  __shared__ float red[8 * 16 * 2];
  __shared__ float stat[16 * 2];
  __shared__ unsigned char shMask[16];

  int tid = threadIdx.x;
  int wave = tid >> 5, lane = tid & 31;
  int brow = blockIdx.x * 16;

  const _Float16* W0p = ws + W0P_OFF;
  const _Float16* U0p = ws + U0P_OFF;
  const _Float16* W1p = ws + W1P_OFF;
  const _Float16* U1p = ws + U1P_OFF;

  for (int i = tid; i < 16 * HH; i += 256) {
    sh_h0[i] = (_Float16)0.f; sh_h0a[i] = (_Float16)0.f; sh_h1[i] = (_Float16)0.f;
  }
  const v8f vzero = {0.f,0.f,0.f,0.f,0.f,0.f,0.f,0.f};
  v8f c0[2], c1[2];
  c0[0] = vzero; c0[1] = vzero; c1[0] = vzero; c1[1] = vzero;
  __syncthreads();

  for (int t = 0; t < TT; ++t) {
    // stage x_t (f32 -> f16) + mask_t into LDS
    for (int i = tid; i < 16 * DD; i += 256) {
      int r = i >> 6, d = i & 63;
      sh_x[i] = (_Float16)x[((size_t)(brow + r) * TT + t) * DD + d];
    }
    if (tid < 16) shMask[tid] = mask[(size_t)(brow + tid) * TT + t];
    __syncthreads();

    cell<DD>(sh_x, sh_h0, W0p, U0p, gk0, bk0, gr0, br0, b0v, gs0, bs0,
             shMask, red, stat, zbuf, c0, sh_h0a, sh_h0, wave, lane, tid);
    cell<HH>(sh_h0a, sh_h1, W1p, U1p, gk1, bk1, gr1, br1, b1v, gs1, bs1,
             shMask, red, stat, zbuf, c1, (_Float16*)nullptr, sh_h1, wave, lane, tid);
  }

  // final projection: out = tanh(h1 @ Wd + bd), 16x64 per workgroup
  for (int o = tid; o < 16 * 64; o += 256) {
    int r = o >> 6, zc = o & 63;
    float acc = bd[zc];
    for (int j = 0; j < HH; ++j) acc += (float)sh_h1[r * HH + j] * Wd[j * 64 + zc];
    out[(size_t)(brow + r) * 64 + zc] = tanhf(acc);
  }
}

extern "C" void kernel_launch(void* const* d_in, const int* in_sizes, int n_in,
                              void* d_out, int out_size, void* d_ws, size_t ws_size,
                              hipStream_t stream) {
  (void)in_sizes; (void)n_in; (void)out_size; (void)ws_size;
  const float*         x    = (const float*)d_in[0];
  const unsigned char* mask = (const unsigned char*)d_in[1];
  const float* W0  = (const float*)d_in[2];
  const float* U0  = (const float*)d_in[3];
  const float* b0  = (const float*)d_in[4];
  const float* gk0 = (const float*)d_in[5];
  const float* bk0 = (const float*)d_in[6];
  const float* gr0 = (const float*)d_in[7];
  const float* br0 = (const float*)d_in[8];
  const float* gs0 = (const float*)d_in[9];
  const float* bs0 = (const float*)d_in[10];
  const float* W1  = (const float*)d_in[11];
  const float* U1  = (const float*)d_in[12];
  const float* b1  = (const float*)d_in[13];
  const float* gk1 = (const float*)d_in[14];
  const float* bk1 = (const float*)d_in[15];
  const float* gr1 = (const float*)d_in[16];
  const float* br1 = (const float*)d_in[17];
  const float* gs1 = (const float*)d_in[18];
  const float* bs1 = (const float*)d_in[19];
  const float* Wd  = (const float*)d_in[20];
  const float* bd  = (const float*)d_in[21];
  _Float16* wsp = (_Float16*)d_ws;

  pack_weights<<<(PACK_TOTAL + 255) / 256, 256, 0, stream>>>(W0, U0, W1, U1, wsp);
  lstm_persist<<<2, 256, 0, stream>>>(x, mask, wsp,
                                      gk0, bk0, gr0, br0, b0, gs0, bs0,
                                      gk1, bk1, gr1, br1, b1, gs1, bs1,
                                      Wd, bd, (float*)d_out);
}
